// MetaCA_10488310137253
// MI455X (gfx1250) — matrix operand order repaired and checked by reading, not compiled, not run on hardware
//
#include <hip/hip_runtime.h>
#include <hip/hip_bf16.h>

// ---------------------------------------------------------------------------
// MetaCA on gfx1250: fused rule-MLP evolve via v_wmma_f32_16x16x32_f16
// ---------------------------------------------------------------------------

typedef _Float16 v16h __attribute__((ext_vector_type(16)));
typedef _Float16 v8h  __attribute__((ext_vector_type(8)));
typedef _Float16 v4h  __attribute__((ext_vector_type(4)));
typedef float    v8f  __attribute__((ext_vector_type(8)));

#define BB   8
#define TT   4096
#define DD   128
#define RR   8
#define D2   256
#define D3   384
#define MT   128        // rows (t) per block
#define CLS  136        // cells LDS row stride in halves (128 + 8 pad)
#define HLS  264        // h LDS row stride in halves (256 + 8 pad)
#define NBLK (BB * (TT / MT))   // 256 blocks

__device__ __forceinline__ v8f wmma16(v16h a, v16h b, v8f c) {
  // D = A(16x32,f16) * B(32x16,f16) + C(16x16,f32)
  return __builtin_amdgcn_wmma_f32_16x16x32_f16(false, a, false, b,
                                                (short)0, c, false, false);
}

// A-fragment: lane l (m=l&15, hi=l>>4)
// holds K = {hi*16+0..15} -> one contiguous 32B run (matches B fragments).
__device__ __forceinline__ v16h ld_afrag(const _Float16* p) {
  v8h lo = *(const v8h*)p;
  v8h hv = *(const v8h*)(p + 8);
  v16h a;
#pragma unroll
  for (int e = 0; e < 8; ++e) { a[e] = lo[e]; a[8 + e] = hv[e]; }
  return a;
}

__device__ __forceinline__ float gelu_exact(float x) {
  return 0.5f * x * (1.0f + erff(x * 0.70710678118654752f));
}

// gfx1250 has a native v_tanh_f32 (TRANS op). Use it when clang exposes it.
__device__ __forceinline__ float fast_tanh(float x) {
#if defined(__has_builtin)
#if __has_builtin(__builtin_amdgcn_tanhf)
  return __builtin_amdgcn_tanhf(x);
#else
  return tanhf(x);
#endif
#else
  return tanhf(x);
#endif
}

// ---------------------------------------------------------------------------
// Setup: run the 3 tiny selector MLPs once (rule softmax, n_evolve, alpha)
// ---------------------------------------------------------------------------
__global__ void setup_kernel(const float* __restrict__ c,
                             const float* __restrict__ rW1, const float* __restrict__ rb1,
                             const float* __restrict__ rW2, const float* __restrict__ rb2,
                             const float* __restrict__ sW1, const float* __restrict__ sb1,
                             const float* __restrict__ sW2, const float* __restrict__ sb2,
                             const float* __restrict__ aW1, const float* __restrict__ ab1,
                             const float* __restrict__ aW2, const float* __restrict__ ab2,
                             float* __restrict__ rw_out, float* __restrict__ alpha_out,
                             int* __restrict__ nev_out) {
  __shared__ float hbuf[64];
  __shared__ float lbuf[8];
  const int tid = threadIdx.x;

  // ---- rule selector: 128 -> 64 -> 8, softmax
  if (tid < 64) {
    float s = rb1[tid];
    for (int i = 0; i < 128; ++i) s += c[i] * rW1[i * 64 + tid];
    hbuf[tid] = gelu_exact(s);
  }
  __syncthreads();
  if (tid < 8) {
    float s = rb2[tid];
    for (int j = 0; j < 64; ++j) s += hbuf[j] * rW2[j * 8 + tid];
    lbuf[tid] = s;
  }
  __syncthreads();
  if (tid == 0) {
    float mx = lbuf[0];
    for (int k = 1; k < 8; ++k) mx = fmaxf(mx, lbuf[k]);
    float sum = 0.f, e[8];
    for (int k = 0; k < 8; ++k) { e[k] = expf(lbuf[k] - mx); sum += e[k]; }
    for (int k = 0; k < 8; ++k) rw_out[k] = e[k] / sum;
  }
  __syncthreads();

  // ---- steps selector: 128 -> 32 -> 7, softmax -> rounded expectation
  if (tid < 32) {
    float s = sb1[tid];
    for (int i = 0; i < 128; ++i) s += c[i] * sW1[i * 32 + tid];
    hbuf[tid] = gelu_exact(s);
  }
  __syncthreads();
  if (tid < 7) {
    float s = sb2[tid];
    for (int j = 0; j < 32; ++j) s += hbuf[j] * sW2[j * 7 + tid];
    lbuf[tid] = s;
  }
  __syncthreads();
  if (tid == 0) {
    float mx = lbuf[0];
    for (int k = 1; k < 7; ++k) mx = fmaxf(mx, lbuf[k]);
    float sum = 0.f, e[7];
    for (int k = 0; k < 7; ++k) { e[k] = expf(lbuf[k] - mx); sum += e[k]; }
    float nsoft = 0.f;
    for (int k = 0; k < 7; ++k) nsoft += (e[k] / sum) * (float)(k + 2);
    int ne = (int)(nsoft + 0.5f);      // python int() truncation, nsoft > 0
    ne = ne < 2 ? 2 : (ne > 8 ? 8 : ne);
    *nev_out = ne;
  }
  __syncthreads();

  // ---- residual selector: 128 -> 32 -> 1, sigmoid
  if (tid < 32) {
    float s = ab1[tid];
    for (int i = 0; i < 128; ++i) s += c[i] * aW1[i * 32 + tid];
    hbuf[tid] = gelu_exact(s);
  }
  __syncthreads();
  if (tid == 0) {
    float z = ab2[0];
    for (int j = 0; j < 32; ++j) z += hbuf[j] * aW2[j];
    *alpha_out = 0.1f + 0.8f / (1.0f + expf(-z));
  }
}

// ---------------------------------------------------------------------------
// One-time: transpose + convert weights to f16 (N-major, K-contiguous) so
// WMMA B-fragments are single contiguous 32B loads per lane.
// ---------------------------------------------------------------------------
__global__ void convert_weights(const float* __restrict__ W1, const float* __restrict__ W2,
                                _Float16* __restrict__ W1T, _Float16* __restrict__ W2T) {
  const int n1 = RR * D3 * D2;  // 786432
  const int n2 = RR * D2 * DD;  // 262144
  const int stride = gridDim.x * blockDim.x;
  for (int idx = blockIdx.x * blockDim.x + threadIdx.x; idx < n1; idx += stride) {
    int r = idx / (D3 * D2);
    int rem = idx - r * D3 * D2;
    int k = rem >> 8;           // /256
    int n = rem & 255;
    W1T[((size_t)r * D2 + n) * D3 + k] = (_Float16)W1[idx];
  }
  for (int idx = blockIdx.x * blockDim.x + threadIdx.x; idx < n2; idx += stride) {
    int r = idx / (D2 * DD);
    int rem = idx - r * D2 * DD;
    int k = rem >> 7;           // /128
    int n = rem & 127;
    W2T[((size_t)r * DD + n) * D2 + k] = (_Float16)W2[idx];
  }
}

// ---------------------------------------------------------------------------
// Fused evolve step: concat(x, roll+1, roll-1) @ W1 -> GELU -> @ W2 -> tanh
// -> rule-weighted sum -> alpha blend.   256 threads = 8 wave32.
// ---------------------------------------------------------------------------
__global__ __launch_bounds__(256) void evolve_kernel(
    const float* __restrict__ src, float* __restrict__ dst,
    const _Float16* __restrict__ w1t, const _Float16* __restrict__ w2t,
    const float* __restrict__ b1g, const float* __restrict__ b2g,
    const float* __restrict__ rwp, const float* __restrict__ alphap,
    const int* __restrict__ nevp, int iter) {
  const int tid = threadIdx.x;
  const int batch = blockIdx.x >> 5;          // T/MT == 32 tiles per batch
  const int t0 = (blockIdx.x & 31) * MT;
  const size_t base = ((size_t)batch * TT + t0) * DD;

  if (iter >= *nevp) {  // inactive step: deterministic passthrough keeps parity
    const float4* s4 = (const float4*)(src + base);
    float4* d4 = (float4*)(dst + base);
    for (int i = tid; i < MT * DD / 4; i += 256) d4[i] = s4[i];
    return;
  }

  extern __shared__ char smem[];
  _Float16* cellsLds = (_Float16*)smem;                      // (MT+2) x CLS
  _Float16* hLds = (_Float16*)(smem + (MT + 2) * CLS * 2);   // MT x HLS

  // Stage cells tile + 1-row halo each side (torus wrap along T) as f16.
  // float4 global loads -> packed 4xf16 (8B) DS stores.
  for (int i = tid; i < (MT + 2) * (DD / 4); i += 256) {
    int rr = i >> 5;                 // / (DD/4)
    int c4 = (i & 31) << 2;
    int t = (t0 + rr - 1) & (TT - 1);
    float4 f = *(const float4*)(src + ((size_t)batch * TT + t) * DD + c4);
    v4h h4;
    h4[0] = (_Float16)f.x; h4[1] = (_Float16)f.y;
    h4[2] = (_Float16)f.z; h4[3] = (_Float16)f.w;
    *(v4h*)(cellsLds + rr * CLS + c4) = h4;
  }
  __syncthreads();

  const int wave = tid >> 5;
  const int lane = tid & 31;
  const int m16 = lane & 15;
  const int hi = lane >> 4;

  const v8f vzero = {};
  v8f acc[8];
#pragma unroll
  for (int i = 0; i < 8; ++i) acc[i] = vzero;

#pragma unroll 1
  for (int r = 0; r < RR; ++r) {
    // Hoist per-rule scalars so these loads overlap the GEMM1 wmma phase.
    const float bias1a = b1g[r * D2 + 32 * wave + m16];
    const float bias1b = b1g[r * D2 + 32 * wave + 16 + m16];
    const float bias2 = b2g[r * DD + 16 * wave + m16];
    const float wr = rwp[r];

    // ---------------- GEMM1: [128 x 384] @ W1T_r -> h [128 x 256] ----------
    // Wave owns N-tiles {2w, 2w+1}; loops all 8 M-strips (B reused 8x).
    v8f c1[2][8];
#pragma unroll
    for (int i = 0; i < 2; ++i)
#pragma unroll
      for (int j = 0; j < 8; ++j) c1[i][j] = vzero;

    const _Float16* w1r = w1t + (size_t)r * D2 * D3;
#pragma unroll 1
    for (int kt = 0; kt < 12; ++kt) {
      const int region = kt >> 2;                 // 0:self 1:left(t-1) 2:right(t+1)
      const int kk = (kt & 3) << 5;               // K within the 128-wide region
      const int rowoff = (region == 0) ? 1 : ((region == 1) ? 0 : 2);
      const int kcol = region * DD + kk;          // K index into W1 (0..383)
      v16h bf0 = *(const v16h*)(w1r + (size_t)(32 * wave + m16) * D3 + kcol + hi * 16);
      v16h bf1 = *(const v16h*)(w1r + (size_t)(32 * wave + 16 + m16) * D3 + kcol + hi * 16);
#pragma unroll
      for (int ms = 0; ms < 8; ++ms) {
        v16h a = ld_afrag(cellsLds + (16 * ms + m16 + rowoff) * CLS + kk + hi * 16);
        c1[0][ms] = wmma16(a, bf0, c1[0][ms]);
        c1[1][ms] = wmma16(a, bf1, c1[1][ms]);
      }
    }

    // bias + exact GELU -> f16 h tile in LDS
#pragma unroll
    for (int cc = 0; cc < 2; ++cc) {
      const int ncol = 32 * wave + 16 * cc + m16;
      const float bias = cc ? bias1b : bias1a;
#pragma unroll
      for (int ms = 0; ms < 8; ++ms) {
        const int rowb = 16 * ms + 8 * hi;
#pragma unroll
        for (int v = 0; v < 8; ++v) {
          hLds[(rowb + v) * HLS + ncol] = (_Float16)gelu_exact(c1[cc][ms][v] + bias);
        }
      }
    }
    __syncthreads();

    // ---------------- GEMM2: h [128 x 256] @ W2T_r -> [128 x 128] ----------
    v8f c2[8];
#pragma unroll
    for (int j = 0; j < 8; ++j) c2[j] = vzero;

    const _Float16* w2r = w2t + (size_t)r * DD * D2;
#pragma unroll 1
    for (int kt = 0; kt < 8; ++kt) {
      const int kb = kt << 5;
      v16h bf = *(const v16h*)(w2r + (size_t)(16 * wave + m16) * D2 + kb + hi * 16);
#pragma unroll
      for (int ms = 0; ms < 8; ++ms) {
        v16h a = ld_afrag(hLds + (16 * ms + m16) * HLS + kb + hi * 16);
        c2[ms] = wmma16(a, bf, c2[ms]);
      }
    }
    __syncthreads();  // h LDS free for next rule

#pragma unroll
    for (int ms = 0; ms < 8; ++ms)
#pragma unroll
      for (int v = 0; v < 8; ++v) acc[ms][v] += wr * fast_tanh(c2[ms][v] + bias2);
  }

  // alpha blend against f32 source state
  const float al = *alphap;
  const float be = 1.0f - al;
  const int col = 16 * wave + m16;
#pragma unroll
  for (int ms = 0; ms < 8; ++ms) {
    const int rowb = 16 * ms + 8 * hi;
#pragma unroll
    for (int v = 0; v < 8; ++v) {
      const size_t off = base + (size_t)(rowb + v) * DD + col;
      dst[off] = al * src[off] + be * acc[ms][v];
    }
  }
}

// ---------------------------------------------------------------------------
// LayerNorm over D=128, one wave32 per row.
// ---------------------------------------------------------------------------
__global__ __launch_bounds__(256) void ln_kernel(const float* __restrict__ x,
                                                 const float* __restrict__ g,
                                                 const float* __restrict__ b,
                                                 float* __restrict__ out) {
  const int row = blockIdx.x * 8 + (threadIdx.x >> 5);
  const int lane = threadIdx.x & 31;
  const float* xr = x + (size_t)row * DD;
  float v[4], s = 0.f, sq = 0.f;
#pragma unroll
  for (int j = 0; j < 4; ++j) {
    v[j] = xr[lane + 32 * j];
    s += v[j];
    sq += v[j] * v[j];
  }
#pragma unroll
  for (int off = 16; off > 0; off >>= 1) {
    s += __shfl_xor(s, off, 32);
    sq += __shfl_xor(sq, off, 32);
  }
  const float mean = s * (1.0f / DD);
  const float var = sq * (1.0f / DD) - mean * mean;
  const float inv = rsqrtf(var + 1e-5f);
#pragma unroll
  for (int j = 0; j < 4; ++j) {
    const int c = lane + 32 * j;
    out[(size_t)row * DD + c] = (v[j] - mean) * inv * g[c] + b[c];
  }
}

// ---------------------------------------------------------------------------
extern "C" void kernel_launch(void* const* d_in, const int* in_sizes, int n_in,
                              void* d_out, int out_size, void* d_ws, size_t ws_size,
                              hipStream_t stream) {
  const float* cells = (const float*)d_in[0];
  const float* c_state = (const float*)d_in[1];
  const float* W1 = (const float*)d_in[2];
  const float* b1 = (const float*)d_in[3];
  const float* W2 = (const float*)d_in[4];
  const float* b2 = (const float*)d_in[5];
  const float* rsel_W1 = (const float*)d_in[6];
  const float* rsel_b1 = (const float*)d_in[7];
  const float* rsel_W2 = (const float*)d_in[8];
  const float* rsel_b2 = (const float*)d_in[9];
  const float* ssel_W1 = (const float*)d_in[10];
  const float* ssel_b1 = (const float*)d_in[11];
  const float* ssel_W2 = (const float*)d_in[12];
  const float* ssel_b2 = (const float*)d_in[13];
  const float* asel_W1 = (const float*)d_in[14];
  const float* asel_b1 = (const float*)d_in[15];
  const float* asel_W2 = (const float*)d_in[16];
  const float* asel_b2 = (const float*)d_in[17];
  const float* ln_g = (const float*)d_in[18];
  const float* ln_b = (const float*)d_in[19];

  // Workspace layout (all offsets 64B-aligned):
  //  [0)     float rw[8]
  //  [32)    float alpha
  //  [36)    int   n_evolve
  //  [64)    f16 W1T  [R][256][384]   1,572,864 B
  //  [..)    f16 W2T  [R][128][256]     524,288 B
  //  [..)    f32 bufA [B][T][D]      16,777,216 B
  //  [..)    f32 bufB [B][T][D]      16,777,216 B
  char* ws = (char*)d_ws;
  float* rw = (float*)ws;
  float* alpha = (float*)(ws + 32);
  int* nev = (int*)(ws + 36);
  _Float16* W1T = (_Float16*)(ws + 64);
  _Float16* W2T = W1T + (size_t)RR * D2 * D3;
  float* bufA = (float*)(ws + 64 + ((size_t)RR * D2 * D3 + (size_t)RR * DD * D2) * 2);
  float* bufB = bufA + (size_t)BB * TT * DD;

  setup_kernel<<<1, 128, 0, stream>>>(c_state,
                                      rsel_W1, rsel_b1, rsel_W2, rsel_b2,
                                      ssel_W1, ssel_b1, ssel_W2, ssel_b2,
                                      asel_W1, asel_b1, asel_W2, asel_b2,
                                      rw, alpha, nev);
  convert_weights<<<3072, 256, 0, stream>>>(W1, W2, W1T, W2T);

  const size_t SMEM = (size_t)(MT + 2) * CLS * 2 + (size_t)MT * HLS * 2;  // ~101 KB
  for (int it = 0; it < 8; ++it) {
    const float* s = (it == 0) ? cells : ((it & 1) ? bufA : bufB);
    float* d = (it & 1) ? bufB : bufA;
    evolve_kernel<<<NBLK, 256, SMEM, stream>>>(s, d, W1T, W2T, b1, b2, rw, alpha, nev, it);
  }

  ln_kernel<<<BB * TT / 8, 256, 0, stream>>>(bufB, ln_g, ln_b, (float*)d_out);
}
